// GAT_46961172415075
// MI455X (gfx1250) — matrix operand order repaired and checked by reading, not compiled, not run on hardware
//
#include <hip/hip_runtime.h>
#include <hip/hip_bf16.h>

// ---------------------------------------------------------------------------
// GAT forward for MI455X (gfx1250).
//  - Dense GEMMs via v_wmma_f32_16x16x32_bf16 with a bf16 hi/lo split
//    (3 WMMAs per tile) for fp32-grade accuracy.
//  - Edge softmax/aggregation via L2-resident f32 atomics (192MB L2 holds the
//    entire per-layer working set, so random gather/scatter never hits HBM).
// ---------------------------------------------------------------------------

typedef __attribute__((ext_vector_type(16))) __bf16          v16bf;
typedef __attribute__((ext_vector_type(8)))  float           v8f;
typedef __attribute__((ext_vector_type(8)))  unsigned short  v8u;
typedef __attribute__((ext_vector_type(16))) unsigned short  v16u;

#define KPAD   128   // K padded (max real K = 128, layer0 K=100 zero-padded)
#define NSTR   256   // GEMM output column stride (fout + sout padded)
#define ASTR   192   // aggregation buffer column stride (max fout = 188)
#define HEADS  4

__device__ __forceinline__ unsigned short f2bf(float f) {
    unsigned int u = __float_as_uint(f);
    u += 0x7FFFu + ((u >> 16) & 1u);          // round-to-nearest-even
    return (unsigned short)(u >> 16);
}

__device__ __forceinline__ float bf2f(unsigned short h) {
    return __uint_as_float(((unsigned int)h) << 16);
}

__device__ __forceinline__ v16bf ld16bf(const unsigned short* p0,
                                        const unsigned short* p1) {
    v8u a = *(const v8u*)p0;
    v8u b = *(const v8u*)p1;
    v16u u = __builtin_shufflevector(a, b, 0,1,2,3,4,5,6,7,8,9,10,11,12,13,14,15);
    return __builtin_bit_cast(v16bf, u);
}

__device__ __forceinline__ void fAtomicAdd(float* p, float v) {
    __hip_atomic_fetch_add(p, v, __ATOMIC_RELAXED, __HIP_MEMORY_SCOPE_AGENT);
}

// float atomic max via signed-max / unsigned-min trick (works for mixed signs)
__device__ __forceinline__ void fAtomicMax(float* p, float v) {
    if (v >= 0.0f)
        __hip_atomic_fetch_max((int*)p, __float_as_int(v),
                               __ATOMIC_RELAXED, __HIP_MEMORY_SCOPE_AGENT);
    else
        __hip_atomic_fetch_min((unsigned int*)p, __float_as_uint(v),
                               __ATOMIC_RELAXED, __HIP_MEMORY_SCOPE_AGENT);
}

// --------------------------- prep kernels ----------------------------------

// x [N,100] f32 -> A_hi/A_lo [N,128] bf16 (zero-padded K)
__global__ void prep_x(const float* __restrict__ x,
                       unsigned short* __restrict__ Ahi,
                       unsigned short* __restrict__ Alo, int N, int K) {
    int i = blockIdx.x * blockDim.x + threadIdx.x;
    if (i >= N * KPAD) return;
    int n = i >> 7, k = i & (KPAD - 1);
    float v = (k < K) ? x[(size_t)n * K + k] : 0.0f;
    unsigned short hi = f2bf(v);
    Ahi[i] = hi;
    Alo[i] = f2bf(v - bf2f(hi));
}

// Fuse W [fout,K] and sw [sout,K] into Bcat_hi/lo [NSTR,KPAD] bf16
__global__ void prep_B(const float* __restrict__ W, const float* __restrict__ sw,
                       unsigned short* __restrict__ Bhi,
                       unsigned short* __restrict__ Blo,
                       int fout, int sout, int K) {
    int i = blockIdx.x * blockDim.x + threadIdx.x;
    if (i >= NSTR * KPAD) return;
    int col = i >> 7, k = i & (KPAD - 1);
    float v = 0.0f;
    if (k < K) {
        if (col < fout)              v = W[(size_t)col * K + k];
        else if (col < fout + sout)  v = sw[(size_t)(col - fout) * K + k];
    }
    unsigned short hi = f2bf(v);
    Bhi[i] = hi;
    Blo[i] = f2bf(v - bf2f(hi));
}

// --------------------------- WMMA GEMM -------------------------------------
// hcat[N, NSTR] = A[N, KPAD] * Bcat[NSTR, KPAD]^T  (bf16x3, f32 accumulate)
// One wave computes a 16x64 strip; block = 4 waves = 16x256; grid = N/16.
__global__ __launch_bounds__(128) void gemm_bf16x3(
        const unsigned short* __restrict__ Ahi,
        const unsigned short* __restrict__ Alo,
        const unsigned short* __restrict__ Bhi,
        const unsigned short* __restrict__ Blo,
        float* __restrict__ Ccat) {
    const int lane  = threadIdx.x & 31;
    const int wave  = threadIdx.x >> 5;       // N quadrant: 64 cols each
    const int mBase = blockIdx.x * 16;
    const int half  = lane >> 4;              // 0 | 1
    const int l15   = lane & 15;

    // Operand k order used here (16-bit, wave32; contiguous, not ISA table order):
    //  A 16x32: row = l15; lane<16 -> K 0..15 ; lane>=16 -> K 16..31
    //  B 32x16: col = l15; lane<16 -> K 0..15 ; lane>=16 -> K 16..31
    const int aSeg = half * 16;
    const int bK   = half * 16;
    const size_t aRowOff = (size_t)(mBase + l15) * KPAD;

    v8f acc[4] = {};
#pragma unroll
    for (int kt = 0; kt < KPAD / 32; ++kt) {
        const int k0 = kt * 32;
        const unsigned short* ah = Ahi + aRowOff + k0 + aSeg;
        const unsigned short* al = Alo + aRowOff + k0 + aSeg;
        v16bf a_hi = ld16bf(ah, ah + 8);
        v16bf a_lo = ld16bf(al, al + 8);
#pragma unroll
        for (int j = 0; j < 4; ++j) {
            const size_t bRowOff = (size_t)(wave * 64 + j * 16 + l15) * KPAD;
            const unsigned short* bh = Bhi + bRowOff + k0 + bK;
            const unsigned short* bl = Blo + bRowOff + k0 + bK;
            v16bf b_hi = ld16bf(bh, bh + 8);
            v16bf b_lo = ld16bf(bl, bl + 8);
            // D = Ahi*Bhi + Ahi*Blo + Alo*Bhi  (fp32-grade via bf16x3)
            acc[j] = __builtin_amdgcn_wmma_f32_16x16x32_bf16(
                         false, a_hi, false, b_hi, (short)0, acc[j], false, false);
            acc[j] = __builtin_amdgcn_wmma_f32_16x16x32_bf16(
                         false, a_hi, false, b_lo, (short)0, acc[j], false, false);
            acc[j] = __builtin_amdgcn_wmma_f32_16x16x32_bf16(
                         false, a_lo, false, b_hi, (short)0, acc[j], false, false);
        }
    }
    // D layout: VGPR r -> M = r + 8*half, N = l15
    const int rowOff = half * 8;
#pragma unroll
    for (int j = 0; j < 4; ++j) {
        const int col = wave * 64 + j * 16 + l15;
#pragma unroll
        for (int r = 0; r < 8; ++r)
            Ccat[(size_t)(mBase + r + rowOff) * NSTR + col] = acc[j][r];
    }
}

// --------------------------- attention kernels -----------------------------

// per (node, head): al = <h, a_s>, ar = <h, a_d>; init emax/denom
__global__ void node_scores(const float* __restrict__ hcat,
                            const float* __restrict__ a_s,
                            const float* __restrict__ a_d,
                            float* __restrict__ al, float* __restrict__ ar,
                            float* __restrict__ emax, float* __restrict__ denom,
                            int N, int C) {
    int i = blockIdx.x * blockDim.x + threadIdx.x;
    if (i >= N * HEADS) return;
    int n = i / HEADS, h = i % HEADS;
    const float* hp = hcat + (size_t)n * NSTR + h * C;
    float sa = 0.0f, sd = 0.0f;
    for (int c = 0; c < C; ++c) {
        float v = hp[c];
        sa += v * a_s[h * C + c];
        sd += v * a_d[h * C + c];
    }
    al[i] = sa; ar[i] = sd;
    emax[i]  = -__builtin_inff();
    denom[i] = 0.0f;
}

__global__ void zero_f32(float* __restrict__ p, long n) {
    long i = (long)blockIdx.x * blockDim.x + threadIdx.x;
    if (i < n) p[i] = 0.0f;
}

// pass 1: e = leakyrelu(al[src]+ar[dst]); store e; segment max into emax[dst]
__global__ void edge_max(const int* __restrict__ ei, int E, int N,
                         const float* __restrict__ al, const float* __restrict__ ar,
                         float* __restrict__ eb, float* __restrict__ emax) {
    int t = blockIdx.x * blockDim.x + threadIdx.x;
    int ET = E + N;
    if (t >= ET) return;
    int s = (t < E) ? ei[t]     : (t - E);   // self loop tail
    int d = (t < E) ? ei[E + t] : (t - E);
#pragma unroll
    for (int h = 0; h < HEADS; ++h) {
        float e = al[s * HEADS + h] + ar[d * HEADS + h];
        e = (e > 0.0f) ? e : 0.2f * e;
        eb[(size_t)t * HEADS + h] = e;
        fAtomicMax(&emax[d * HEADS + h], e);
    }
}

// pass 2: ee = exp(e - emax[dst]); store; segment sum into denom[dst]
__global__ void edge_expsum(const int* __restrict__ ei, int E, int N,
                            float* __restrict__ eb,
                            const float* __restrict__ emax,
                            float* __restrict__ denom) {
    int t = blockIdx.x * blockDim.x + threadIdx.x;
    int ET = E + N;
    if (t >= ET) return;
    int d = (t < E) ? ei[E + t] : (t - E);
#pragma unroll
    for (int h = 0; h < HEADS; ++h) {
        float ee = __expf(eb[(size_t)t * HEADS + h] - emax[d * HEADS + h]);
        eb[(size_t)t * HEADS + h] = ee;
        fAtomicAdd(&denom[d * HEADS + h], ee);
    }
}

// pass 3: agg[dst] += (ee/denom[dst]) * h[src]   (one thread per edge*head)
__global__ void edge_agg(const int* __restrict__ ei, int E, int N,
                         const float* __restrict__ eb,
                         const float* __restrict__ denom,
                         const float* __restrict__ hcat,
                         float* __restrict__ agg, int C) {
    int idx = blockIdx.x * blockDim.x + threadIdx.x;
    int ET4 = (E + N) * HEADS;
    if (idx >= ET4) return;
    int t = idx >> 2, h = idx & 3;
    int s = (t < E) ? ei[t]     : (t - E);
    int d = (t < E) ? ei[E + t] : (t - E);
    float alpha = eb[idx] / denom[d * HEADS + h];
    const float* hs = hcat + (size_t)s * NSTR + h * C;
    float*       ag = agg  + (size_t)d * ASTR + h * C;
    for (int c = 0; c < C; ++c)
        fAtomicAdd(&ag[c], alpha * hs[c]);
}

// layers 0/1 epilogue: x' = ELU(agg + b + skip + sb); emit bf16 hi/lo for next GEMM
__global__ void finalize01(const float* __restrict__ agg,
                           const float* __restrict__ hcat,
                           const float* __restrict__ b, const float* __restrict__ sb,
                           unsigned short* __restrict__ Ahi,
                           unsigned short* __restrict__ Alo,
                           int N, int fout) {
    int i = blockIdx.x * blockDim.x + threadIdx.x;
    if (i >= N * KPAD) return;               // fout == 128 == KPAD here
    int n = i >> 7, c = i & (KPAD - 1);
    float v = agg[(size_t)n * ASTR + c] + b[c]
            + hcat[(size_t)n * NSTR + fout + c] + sb[c];
    v = (v > 0.0f) ? v : expm1f(v);          // ELU
    unsigned short hi = f2bf(v);
    Ahi[i] = hi;
    Alo[i] = f2bf(v - bf2f(hi));
}

// last layer: head-mean + b + skip + sb, then log_softmax over 47 classes
__global__ void finalize2(const float* __restrict__ agg,
                          const float* __restrict__ hcat,
                          const float* __restrict__ b, const float* __restrict__ sb,
                          float* __restrict__ out, int N) {
    int n = blockIdx.x * blockDim.x + threadIdx.x;
    if (n >= N) return;
    const int C = 47, fout = 188;
    float v[47];
    float mx = -__builtin_inff();
    const float* ag = agg + (size_t)n * ASTR;
    const float* sk = hcat + (size_t)n * NSTR + fout;
    for (int c = 0; c < C; ++c) {
        float m = 0.25f * (ag[c] + ag[C + c] + ag[2 * C + c] + ag[3 * C + c]);
        m += b[c] + sk[c] + sb[c];
        v[c] = m;
        mx = fmaxf(mx, m);
    }
    float s = 0.0f;
    for (int c = 0; c < C; ++c) s += __expf(v[c] - mx);
    float ls = __logf(s) + mx;
    for (int c = 0; c < C; ++c) out[(size_t)n * C + c] = v[c] - ls;
}

// --------------------------- host side -------------------------------------

extern "C" void kernel_launch(void* const* d_in, const int* in_sizes, int n_in,
                              void* d_out, int out_size, void* d_ws, size_t ws_size,
                              hipStream_t stream) {
    const int IN = 100;
    const int N = in_sizes[0] / IN;          // 50000 (multiple of 16)
    const int E = in_sizes[1] / 2;           // 800000
    const int ET = E + N;

    const float* x  = (const float*)d_in[0];
    const int*   ei = (const int*)d_in[1];

    // workspace carve-out (256B aligned)
    char* p = (char*)d_ws;
    auto carve = [&](size_t bytes) -> void* {
        void* r = (void*)p;
        p += (bytes + 255) & ~(size_t)255;
        return r;
    };
    unsigned short* Ahi  = (unsigned short*)carve((size_t)N * KPAD * 2);
    unsigned short* Alo  = (unsigned short*)carve((size_t)N * KPAD * 2);
    unsigned short* Bhi  = (unsigned short*)carve((size_t)NSTR * KPAD * 2);
    unsigned short* Blo  = (unsigned short*)carve((size_t)NSTR * KPAD * 2);
    float* hcat  = (float*)carve((size_t)N * NSTR * 4);
    float* agg   = (float*)carve((size_t)N * ASTR * 4);
    float* al    = (float*)carve((size_t)N * HEADS * 4);
    float* ar    = (float*)carve((size_t)N * HEADS * 4);
    float* emax  = (float*)carve((size_t)N * HEADS * 4);
    float* denom = (float*)carve((size_t)N * HEADS * 4);
    float* eb    = (float*)carve((size_t)ET * HEADS * 4);

    prep_x<<<(N * KPAD + 255) / 256, 256, 0, stream>>>(x, Ahi, Alo, N, IN);

    struct LayerCfg { int base, K, fout, C, sout; };
    const LayerCfg L[3] = {
        { 2, 100, 128, 32, 128},
        { 8, 128, 128, 32, 128},
        {14, 128, 188, 47,  47},
    };

    for (int l = 0; l < 3; ++l) {
        const float* W   = (const float*)d_in[L[l].base + 0];
        const float* a_s = (const float*)d_in[L[l].base + 1];
        const float* a_d = (const float*)d_in[L[l].base + 2];
        const float* b   = (const float*)d_in[L[l].base + 3];
        const float* sw  = (const float*)d_in[L[l].base + 4];
        const float* sb  = (const float*)d_in[L[l].base + 5];

        prep_B<<<(NSTR * KPAD + 255) / 256, 256, 0, stream>>>(
            W, sw, Bhi, Blo, L[l].fout, L[l].sout, L[l].K);

        gemm_bf16x3<<<N / 16, 128, 0, stream>>>(Ahi, Alo, Bhi, Blo, hcat);

        node_scores<<<(N * HEADS + 255) / 256, 256, 0, stream>>>(
            hcat, a_s, a_d, al, ar, emax, denom, N, L[l].C);
        zero_f32<<<((long)N * ASTR + 255) / 256, 256, 0, stream>>>(agg, (long)N * ASTR);

        edge_max<<<(ET + 255) / 256, 256, 0, stream>>>(ei, E, N, al, ar, eb, emax);
        edge_expsum<<<(ET + 255) / 256, 256, 0, stream>>>(ei, E, N, eb, emax, denom);
        edge_agg<<<(ET * HEADS + 255) / 256, 256, 0, stream>>>(
            ei, E, N, eb, denom, hcat, agg, L[l].C);

        if (l < 2)
            finalize01<<<(N * KPAD + 255) / 256, 256, 0, stream>>>(
                agg, hcat, b, sb, Ahi, Alo, N, L[l].fout);
        else
            finalize2<<<(N + 255) / 256, 256, 0, stream>>>(
                agg, hcat, b, sb, (float*)d_out, N);
    }
}
